// ScaledDotProductAttention_28484223107489
// MI455X (gfx1250) — matrix operand order repaired and checked by reading, not compiled, not run on hardware
//
#include <hip/hip_runtime.h>
#include <hip/hip_bf16.h>
#include <math.h>

// ---------------------------------------------------------------------------
// Scaled dot-product attention, MI455X (gfx1250), wave32 + WMMA.
// B=16, S=2048, D=64.  out = [P=softmax(mask(QK^T/8))] @ V ; also emit P.
// Bound by the 268MB attn write -> f16 WMMA w/ f32 accum is free; scores and
// softmax stay on-chip in the 320KB CDNA5 LDS; attn streamed with NT stores.
// ---------------------------------------------------------------------------

#define BATCH 16
#define SEQ   2048
#define DIM   64
#define SROW  2052   // padded f32 row stride for score LDS (bank-conflict free)
#define QROW  66     // padded f16 row stride for Q LDS
#define OROW  68     // padded f32 row stride for O-reduction LDS

typedef __attribute__((ext_vector_type(16))) _Float16 v16h;
typedef __attribute__((ext_vector_type(8)))  float    v8f;
typedef __attribute__((ext_vector_type(4)))  float    f32x4;
typedef __attribute__((ext_vector_type(2)))  float    f32x2;

union V16H { v16h v; _Float16 h[16]; unsigned u[8]; };

__device__ __forceinline__ float wave_max32(float x) {
  #pragma unroll
  for (int o = 16; o > 0; o >>= 1) x = fmaxf(x, __shfl_xor(x, o, 32));
  return x;
}
__device__ __forceinline__ float wave_sum32(float x) {
  #pragma unroll
  for (int o = 16; o > 0; o >>= 1) x += __shfl_xor(x, o, 32);
  return x;
}

// dynamic LDS: scores 16*SROW f32 | Q 16*QROW f16 | rinv 16 f32 | mask SEQ u8
#define SOFF_Q    (16 * SROW * 4)
#define SOFF_RINV (SOFF_Q + 16 * QROW * 2)
#define SOFF_MASK (SOFF_RINV + 16 * 4)
#define SMEM_BYTES (SOFF_MASK + SEQ)

__global__ __launch_bounds__(256)
void sdpa_gfx1250_kernel(const float* __restrict__ Q,
                         const float* __restrict__ K,
                         const float* __restrict__ V,
                         const unsigned char* __restrict__ Mask,
                         float* __restrict__ Out,
                         float* __restrict__ Attn) {
  extern __shared__ char smem[];
  float*         sS    = (float*)smem;                       // 16 x SROW
  _Float16*      sQ    = (_Float16*)(smem + SOFF_Q);         // 16 x QROW
  float*         sRinv = (float*)(smem + SOFF_RINV);         // 16
  unsigned char* sM    = (unsigned char*)(smem + SOFF_MASK); // SEQ bytes

  const int tid  = threadIdx.x;
  const int wave = tid >> 5;
  const int lane = tid & 31;
  const int half = lane >> 4;
  const int l16  = lane & 15;

  const int b  = blockIdx.x >> 7;          // SEQ/16 = 128 q-tiles per batch
  const int q0 = (blockIdx.x & 127) << 4;

  const float* Qb = Q + ((size_t)b * SEQ + q0) * DIM;
  const float* Kb = K + (size_t)b * SEQ * DIM;
  const float* Vb = V + (size_t)b * SEQ * DIM;
  const unsigned char* Mb = Mask + (size_t)b * SEQ;
  float* attnb = Attn + ((size_t)b * SEQ + q0) * SEQ;

  // ---- Phase 0: Q tile -> LDS (f16); mask row -> LDS (coalesced 8B/thr) ----
  {
    const int f = tid * 4;                      // 256 thr * 4 = 16*64
    const f32x4 qv = *(const f32x4*)(Qb + f);
    const int r = f >> 6, c = f & 63;
    _Float16* dst = sQ + r * QROW + c;
    dst[0] = (_Float16)qv.x; dst[1] = (_Float16)qv.y;
    dst[2] = (_Float16)qv.z; dst[3] = (_Float16)qv.w;
    // 256 threads * 8B = 2048B mask row
    *(unsigned long long*)(sM + tid * 8) =
        *(const unsigned long long*)(Mb + tid * 8);
  }
  __syncthreads();

  // ---- Build Q A-operands (16x32 f16, two chunks over D=64) ----
  // A order used here (matches B): row M = lane%16; vgpr j: K = base+16*half+2j; vgpr 4+j: +8.
  V16H aQ[2];
  #pragma unroll
  for (int ch = 0; ch < 2; ++ch) {
    const _Float16* qrow = sQ + l16 * QROW + ch * 32 + 16 * half;
    #pragma unroll
    for (int j = 0; j < 4; ++j) {
      aQ[ch].u[j]     = *(const unsigned*)(qrow + 2 * j);       // K: +2j,+2j+1
      aQ[ch].u[4 + j] = *(const unsigned*)(qrow + 8 + 2 * j);  // K: +8+2j
    }
  }

  // ---- Phase 1: scores S = mask(Q K^T / 8) -> LDS f32 ----
  // Each wave owns 16 consecutive 16-column tiles (covers all 2048 keys).
  for (int tt = 0; tt < 16; ++tt) {
    const int n0 = (wave * 16 + tt) << 4;
    const float* kcol = Kb + (size_t)(n0 + l16) * DIM;  // key row for lane's N
    v8f c = {};
    #pragma unroll
    for (int ch = 0; ch < 2; ++ch) {
      // B order used here (32x16, matches A): col N = lane%16; vgpr j: K = 16*half + 2j, 2j+1.
      V16H bK;
      #pragma unroll
      for (int j = 0; j < 8; ++j) {
        const f32x2 kf = *(const f32x2*)(kcol + ch * 32 + 16 * half + 2 * j);
        bK.h[2 * j]     = (_Float16)kf.x;
        bK.h[2 * j + 1] = (_Float16)kf.y;
      }
      c = __builtin_amdgcn_wmma_f32_16x16x32_f16(false, aQ[ch].v, false, bK.v,
                                                 (short)0, c, false, false);
    }
    // mask depends on key column only; bias: 0 or -inf, applied via 1 FMA
    const float mbias = sM[n0 + l16] ? -__builtin_inff() : 0.f;
    #pragma unroll
    for (int r = 0; r < 8; ++r) {            // C layout: M = r + 8*half, N=l16
      const float sc = fmaf(c[r], 0.125f, mbias);   // finite c: -inf if masked
      sS[(r + 8 * half) * SROW + n0 + l16] = sc;
    }
  }
  __syncthreads();

  // ---- Phase 2: row softmax stats; overwrite LDS with unnormalized e ----
  #pragma unroll
  for (int rr = 0; rr < 2; ++rr) {
    const int row = wave * 2 + rr;
    float* srow = sS + row * SROW;
    float mx = -__builtin_inff();
    for (int i = lane; i < SEQ; i += 32) mx = fmaxf(mx, srow[i]);
    mx = wave_max32(mx);
    float sum = 0.f;
    for (int i = lane; i < SEQ; i += 32) {
      const float s = srow[i];
      const float e = (s == -__builtin_inff()) ? 0.f : __expf(s - mx);
      srow[i] = e;
      sum += e;
    }
    sum = wave_sum32(sum);
    if (lane == 0) sRinv[row] = (sum > 0.f) ? 1.f / sum : 0.f; // NaN-fix -> 0
  }
  __syncthreads();

  // ---- Phase 3a: stream attn = e * rinv to HBM (non-temporal, coalesced) ----
  for (int it = 0; it < 32; ++it) {                 // 32*256*4 = 16*2048
    const int f   = (it * 256 + tid) * 4;
    const int row = f >> 11, col = f & 2047;
    const f32x4 e4 = *(const f32x4*)&sS[row * SROW + col];
    const float rv = sRinv[row];
    const f32x4 p4 = { e4.x * rv, e4.y * rv, e4.z * rv, e4.w * rv };
    __builtin_nontemporal_store(p4, (f32x4*)&attnb[(size_t)row * SEQ + col]);
  }

  // ---- Phase 3b: O_partial = e_slab @ V_slab  (256 keys per wave) ----
  v8f o[4] = {};                      // 16x64 f32 accum (4 N-tiles of 16)
  const int kb0 = wave * 256;
  for (int kc = 0; kc < 8; ++kc) {    // 32-key chunks
    const int kk0 = kb0 + kc * 32;
    V16H a;                           // e (f32 in LDS) -> f16 A operand
    {
      const float* erow = sS + l16 * SROW + kk0 + 16 * half;
      #pragma unroll
      for (int j = 0; j < 4; ++j) {
        const f32x2 p0 = *(const f32x2*)(erow + 2 * j);
        const f32x2 p1 = *(const f32x2*)(erow + 8 + 2 * j);
        a.h[2 * j]         = (_Float16)p0.x;
        a.h[2 * j + 1]     = (_Float16)p0.y;
        a.h[8 + 2 * j]     = (_Float16)p1.x;
        a.h[8 + 2 * j + 1] = (_Float16)p1.y;
      }
    }
    const float* vrow = Vb + (size_t)(kk0 + 16 * half) * DIM + l16;
    #pragma unroll
    for (int nt = 0; nt < 4; ++nt) {  // output dims d0 = nt*16
      V16H bV;                        // B: (K=16*half+2j, N=l16) = V[k][d0+N]
      #pragma unroll
      for (int j = 0; j < 8; ++j) {
        bV.h[2 * j]     = (_Float16)vrow[(2 * j) * DIM + nt * 16];
        bV.h[2 * j + 1] = (_Float16)vrow[(2 * j + 1) * DIM + nt * 16];
      }
      o[nt] = __builtin_amdgcn_wmma_f32_16x16x32_f16(false, a.v, false, bV.v,
                                                     (short)0, o[nt], false, false);
    }
  }

  // ---- Phase 3c: cross-wave reduction of O through (reused) LDS ----
  __syncthreads();                    // all waves done reading sS (e values)
  float* sO = sS;                     // reuse: 8*16*OROW f32 = 34.8KB
  #pragma unroll
  for (int nt = 0; nt < 4; ++nt)
    #pragma unroll
    for (int r = 0; r < 8; ++r)
      sO[wave * 16 * OROW + (r + 8 * half) * OROW + nt * 16 + l16] = o[nt][r];
  __syncthreads();
  {
    const int f = tid * 4;            // 256 thr * 4 = 16*64 outputs
    const int row = f >> 6, col = f & 63;
    float ax = 0.f, ay = 0.f, az = 0.f, aw = 0.f;
    #pragma unroll
    for (int w = 0; w < 8; ++w) {
      const float* p = sO + w * 16 * OROW + row * OROW + col;
      ax += p[0]; ay += p[1]; az += p[2]; aw += p[3];
    }
    const float rv = sRinv[row];
    const f32x4 r4 = { ax * rv, ay * rv, az * rv, aw * rv };
    *(f32x4*)(Out + ((size_t)b * SEQ + q0 + row) * DIM + col) = r4;
  }
}

extern "C" void kernel_launch(void* const* d_in, const int* in_sizes, int n_in,
                              void* d_out, int out_size, void* d_ws, size_t ws_size,
                              hipStream_t stream) {
  (void)in_sizes; (void)n_in; (void)out_size; (void)d_ws; (void)ws_size;
  const float* q = (const float*)d_in[0];
  const float* k = (const float*)d_in[1];
  const float* v = (const float*)d_in[2];
  const unsigned char* mask = (const unsigned char*)d_in[3];
  float* out  = (float*)d_out;                          // [16,2048,64]
  float* attn = out + (size_t)BATCH * SEQ * DIM;        // [16,2048,2048]

  // 320KB LDS per WGP on CDNA5: allow our ~132KB score buffer.
  hipFuncSetAttribute((const void*)sdpa_gfx1250_kernel,
                      hipFuncAttributeMaxDynamicSharedMemorySize,
                      (int)SMEM_BYTES);

  const dim3 grid(BATCH * (SEQ / 16));   // 2048 workgroups: (batch, q-tile)
  const dim3 block(256);                 // 8 wave32 waves
  sdpa_gfx1250_kernel<<<grid, block, SMEM_BYTES, stream>>>(q, k, v, mask, out, attn);
}
